// AdaptiveMultiBranchMambaFeatureFusion_51668456570918
// MI455X (gfx1250) — compile-verified, matrix operand order checked by reading, not run on hardware
//
#include <hip/hip_runtime.h>
#include <math.h>

#define B_    4
#define C_    256
#define H_    48
#define W_    48
#define NTOK  2304          // H_*W_
#define BN_   9216          // B_*NTOK
#define NB_   4
#define DM_   64
#define DI_   128
#define DS_   16
#define DC_   4
#define DTR_  4
#define NH_   8
#define HD_   32

typedef __attribute__((ext_vector_type(16))) _Float16 v16h;
typedef __attribute__((ext_vector_type(8)))  float    v8f;

// ---------------------------------------------------------------------------
// Branch-free WMMA fragment loader.  f32 row-major (R x K) -> f16 fragment for
// v_wmma_f32_16x16x32_f16 (ISA 7.12.2 16-bit A/B layout: lane&15 = row, lanes
// >=16 take k-offset +8; elements 0..7 -> k=kb+0..7, 8..15 -> k=16+kb+0..7).
// Caller guarantees K % 32 == 0.  Out-of-range rows are CLAMPED, not zeroed:
// the resulting garbage lives only in output rows/cols that the store masks.
// ---------------------------------------------------------------------------
__device__ __forceinline__ v16h load_frag(const float* __restrict__ base, int ld,
                                          int r0, int k0, int Rmax, int lane) {
  int r = r0 + (lane & 15);
  r = (r < Rmax) ? r : (Rmax - 1);
  int kb = (lane >> 4) << 3;
  const float* p = base + (size_t)r * ld + k0 + kb;
  float4 u0 = *(const float4*)(p);
  float4 u1 = *(const float4*)(p + 4);
  float4 u2 = *(const float4*)(p + 16);
  float4 u3 = *(const float4*)(p + 20);
  v16h f;
  f[0]=(_Float16)u0.x;  f[1]=(_Float16)u0.y;  f[2]=(_Float16)u0.z;  f[3]=(_Float16)u0.w;
  f[4]=(_Float16)u1.x;  f[5]=(_Float16)u1.y;  f[6]=(_Float16)u1.z;  f[7]=(_Float16)u1.w;
  f[8]=(_Float16)u2.x;  f[9]=(_Float16)u2.y;  f[10]=(_Float16)u2.z; f[11]=(_Float16)u2.w;
  f[12]=(_Float16)u3.x; f[13]=(_Float16)u3.y; f[14]=(_Float16)u3.z; f[15]=(_Float16)u3.w;
  return f;
}

// ---------------------------------------------------------------------------
// Generic GEMM: C[m,n] = sum_k A[m,k] * W[n,k]  (+bias[n]) (+resid[m,n])
// One wave computes a 32x64 output block: 2 A-frags x 4 B-frags -> 8 wmmas per
// 32-wide K step.  Requires M % 32 == 0 and K % 32 == 0 (true for all calls).
// Batched over blockIdx.z with element strides sA/sW/sC.
// ---------------------------------------------------------------------------
__global__ void __launch_bounds__(32)
k_gemm(const float* __restrict__ A, int lda, long long sA,
       const float* __restrict__ Wt, long long sW,
       const float* __restrict__ bias,
       const float* __restrict__ resid, int ldr,
       float* __restrict__ Cout, int ldc, int coff, long long sC,
       int M, int Nw, int K) {
  int lane = threadIdx.x;
  int z = blockIdx.z;
  const float* Ab = A + (size_t)z * sA;
  const float* Wb = Wt + (size_t)z * sW;
  float* Cb = Cout + (size_t)z * sC;
  int mt  = blockIdx.y * 32;
  int nt0 = blockIdx.x * 64;
  v8f acc0[4] = {};
  v8f acc1[4] = {};
  for (int k0 = 0; k0 < K; k0 += 32) {
    v16h a0 = load_frag(Ab, lda, mt,      k0, M, lane);
    v16h a1 = load_frag(Ab, lda, mt + 16, k0, M, lane);
    #pragma unroll
    for (int t = 0; t < 4; ++t) {
      v16h bt = load_frag(Wb, K, nt0 + t * 16, k0, Nw, lane);
      acc0[t] = __builtin_amdgcn_wmma_f32_16x16x32_f16(false, a0, false, bt, (short)0, acc0[t], false, false);
      acc1[t] = __builtin_amdgcn_wmma_f32_16x16x32_f16(false, a1, false, bt, (short)0, acc1[t], false, false);
    }
  }
  int nl = lane & 15;
  int mrow = mt + ((lane >> 4) << 3);   // C-frag: lanes 0-15 rows v, 16-31 rows v+8
  #pragma unroll
  for (int t = 0; t < 4; ++t) {
    int n = nt0 + t * 16 + nl;
    if (n < Nw) {
      float bv = bias ? bias[n] : 0.f;
      #pragma unroll
      for (int v = 0; v < 8; ++v) {
        int m0 = mrow + v;
        int m1 = m0 + 16;
        float r0 = acc0[t][v] + bv;
        float r1 = acc1[t][v] + bv;
        if (resid) {
          r0 += resid[(size_t)m0 * ldr + n];
          r1 += resid[(size_t)m1 * ldr + n];
        }
        Cb[(size_t)m0 * ldc + coff + n] = r0;
        Cb[(size_t)m1 * ldc + coff + n] = r1;
      }
    }
  }
}

// ---------------------------------------------------------------------------
// LayerNorm from NCHW input -> xt (B,N,C) and xnorm (B,N,C)
// ---------------------------------------------------------------------------
__global__ void __launch_bounds__(256)
k_ln_nchw(const float* __restrict__ x, const float* __restrict__ w,
          const float* __restrict__ bb, float* __restrict__ xt,
          float* __restrict__ xnorm) {
  int row = blockIdx.x, c = threadIdx.x;
  int b = row / NTOK, n = row - b * NTOK;
  float v = x[((size_t)b * C_ + c) * NTOK + n];
  __shared__ float s1[256], s2[256];
  s1[c] = v; s2[c] = v * v;
  __syncthreads();
  for (int o = 128; o > 0; o >>= 1) {
    if (c < o) { s1[c] += s1[c + o]; s2[c] += s2[c + o]; }
    __syncthreads();
  }
  float mu = s1[0] * (1.f / 256.f);
  float var = s2[0] * (1.f / 256.f) - mu * mu;
  float rstd = rsqrtf(var + 1e-5f);
  size_t oi = (size_t)row * C_ + c;
  xt[oi] = v;
  xnorm[oi] = (v - mu) * rstd * w[c] + bb[c];
}

// LayerNorm, channels-last input -> out
__global__ void __launch_bounds__(256)
k_ln_chlast(const float* __restrict__ in, const float* __restrict__ w,
            const float* __restrict__ bb, float* __restrict__ out) {
  int row = blockIdx.x, c = threadIdx.x;
  float v = in[(size_t)row * C_ + c];
  __shared__ float s1[256], s2[256];
  s1[c] = v; s2[c] = v * v;
  __syncthreads();
  for (int o = 128; o > 0; o >>= 1) {
    if (c < o) { s1[c] += s1[c + o]; s2[c] += s2[c + o]; }
    __syncthreads();
  }
  float mu = s1[0] * (1.f / 256.f);
  float rstd = rsqrtf(s2[0] * (1.f / 256.f) - mu * mu + 1e-5f);
  out[(size_t)row * C_ + c] = (v - mu) * rstd * w[c] + bb[c];
}

// LayerNorm + exact GeLU, in place
__global__ void __launch_bounds__(256)
k_ln_gelu(float* __restrict__ io, const float* __restrict__ w,
          const float* __restrict__ bb) {
  int row = blockIdx.x, c = threadIdx.x;
  float v = io[(size_t)row * C_ + c];
  __shared__ float s1[256], s2[256];
  s1[c] = v; s2[c] = v * v;
  __syncthreads();
  for (int o = 128; o > 0; o >>= 1) {
    if (c < o) { s1[c] += s1[c + o]; s2[c] += s2[c + o]; }
    __syncthreads();
  }
  float mu = s1[0] * (1.f / 256.f);
  float rstd = rsqrtf(s2[0] * (1.f / 256.f) - mu * mu + 1e-5f);
  float g = (v - mu) * rstd * w[c] + bb[c];
  io[(size_t)row * C_ + c] = 0.5f * g * (1.f + erff(g * 0.70710678118654752f));
}

// ---------------------------------------------------------------------------
// Mamba: causal depthwise conv (K=4, left pad 3) + SiLU.  xz layout
// [br*B+b][l][256] (first DI_=xm_pre, second=z); out xm [g][l][128].
// ---------------------------------------------------------------------------
__global__ void __launch_bounds__(256)
k_mamba_conv(const float* __restrict__ xz, const float* __restrict__ cw,
             const float* __restrict__ cb, float* __restrict__ xm) {
  long long idx = (long long)blockIdx.x * 256 + threadIdx.x;
  if (idx >= (long long)NB_ * B_ * NTOK * DI_) return;
  int d = (int)(idx & (DI_ - 1));
  long long rest = idx >> 7;
  int l = (int)(rest % NTOK);
  int g = (int)(rest / NTOK);
  int br = g >> 2;
  float acc = cb[br * DI_ + d];
  const float* wp = cw + (size_t)(br * DI_ + d) * DC_;
  #pragma unroll
  for (int k = 0; k < DC_; ++k) {
    int lp = l - (DC_ - 1) + k;
    if (lp >= 0)
      acc += wp[k] * xz[((size_t)g * NTOK + lp) * (2 * DI_) + d];
  }
  acc = acc * (1.f / (1.f + __expf(-acc)));   // SiLU
  xm[((size_t)g * NTOK + l) * DI_ + d] = acc;
}

// dt = softplus(xproj[:, :4] @ dt_w.T + dt_b)
__global__ void __launch_bounds__(256)
k_dtproj(const float* __restrict__ xp, const float* __restrict__ dtw,
         const float* __restrict__ dtb, float* __restrict__ dt) {
  long long idx = (long long)blockIdx.x * 256 + threadIdx.x;
  if (idx >= (long long)NB_ * BN_ * DI_) return;
  int d = (int)(idx & (DI_ - 1));
  long long row = idx >> 7;              // br*BN_ + bn
  int br = (int)(row / BN_);
  const float* r = xp + row * 36;
  const float* w = dtw + (size_t)(br * DI_ + d) * DTR_;
  float a = dtb[br * DI_ + d];
  #pragma unroll
  for (int k = 0; k < DTR_; ++k) a += r[k] * w[k];
  dt[row * DI_ + d] = (a > 20.f) ? a : log1pf(__expf(a));
}

// Selective-scan: one thread per (branch, batch, channel d); h[16] in registers.
__global__ void __launch_bounds__(256)
k_scan(const float* __restrict__ dt, const float* __restrict__ xm,
       const float* __restrict__ xp, const float* __restrict__ Alog,
       const float* __restrict__ Dp, float* __restrict__ y) {
  int tid = blockIdx.x * 256 + threadIdx.x;
  if (tid >= NB_ * B_ * DI_) return;
  int d = tid & (DI_ - 1);
  int g = tid >> 7;                      // br*B + b
  int br = g >> 2;
  float A[DS_], h[DS_];
  #pragma unroll
  for (int s = 0; s < DS_; ++s) {
    A[s] = -__expf(Alog[(size_t)(br * DI_ + d) * DS_ + s]);
    h[s] = 0.f;
  }
  float Dv = Dp[br * DI_ + d];
  size_t rowbase = (size_t)g * NTOK;
  for (int l = 0; l < NTOK; ++l) {
    size_t row = rowbase + l;
    float dtv = dt[row * DI_ + d];
    float xv  = xm[row * DI_ + d];
    const float* pr = xp + row * 36;
    __builtin_prefetch(pr + 36, 0, 0);   // global_prefetch_b8: next token's B/C
    float dx = dtv * xv;
    float acc = 0.f;
    #pragma unroll
    for (int s = 0; s < DS_; ++s) {
      float dA = __expf(dtv * A[s]);
      h[s] = dA * h[s] + dx * pr[4 + s];   // Bm at cols 4..19
      acc += h[s] * pr[20 + s];            // Cm at cols 20..35
    }
    y[row * DI_ + d] = acc + xv * Dv;
  }
}

// y *= silu(z)   (z = xz[..., DI_ + d])
__global__ void __launch_bounds__(256)
k_ygate(float* __restrict__ y, const float* __restrict__ xz) {
  long long idx = (long long)blockIdx.x * 256 + threadIdx.x;
  if (idx >= (long long)NB_ * BN_ * DI_) return;
  int d = (int)(idx & (DI_ - 1));
  long long row = idx >> 7;
  float z = xz[row * (2 * DI_) + DI_ + d];
  y[idx] *= z * (1.f / (1.f + __expf(-z)));
}

// m_out += skip*chunk ; scatter into x_concat (B,N,C)
__global__ void __launch_bounds__(256)
k_mout_skip(float* __restrict__ mout, const float* __restrict__ xnorm,
            const float* __restrict__ skip, float* __restrict__ xcat) {
  long long idx = (long long)blockIdx.x * 256 + threadIdx.x;
  if (idx >= (long long)NB_ * BN_ * DM_) return;
  int d = (int)(idx & (DM_ - 1));
  long long row = idx >> 6;              // br*BN_ + bn
  int br = (int)(row / BN_);
  long long bn = row - (long long)br * BN_;
  float ch = xnorm[bn * C_ + br * DM_ + d];
  float v = mout[idx] + skip[0] * ch;
  mout[idx] = v;
  xcat[bn * C_ + br * DM_ + d] = v;
}

// gating grouped dwconv: (B,C,N) groups=NB -> tmpq (b,i,n)
__global__ void __launch_bounds__(256)
k_qdw(const float* __restrict__ xcat, const float* __restrict__ w,
      const float* __restrict__ bias, float* __restrict__ tmpq) {
  int idx = blockIdx.x * 256 + threadIdx.x;
  if (idx >= B_ * NB_ * NTOK) return;
  int n = idx % NTOK;
  int i = (idx / NTOK) % NB_;
  int b = idx / (NTOK * NB_);
  float acc = bias[i];
  #pragma unroll
  for (int k = 0; k < 3; ++k) {
    int nn = n + k - 1;
    if (nn >= 0 && nn < NTOK) {
      const float* src = xcat + ((size_t)b * NTOK + nn) * C_ + i * DM_;
      for (int d = 0; d < DM_; ++d)
        acc += w[(i * DM_ + d) * 3 + k] * src[d];
    }
  }
  tmpq[((size_t)b * NB_ + i) * NTOK + n] = acc;
}

// pointwise (NB->NB) + sigmoid -> g
__global__ void __launch_bounds__(256)
k_qpw(const float* __restrict__ tmpq, const float* __restrict__ w,
      const float* __restrict__ bias, float* __restrict__ g) {
  int idx = blockIdx.x * 256 + threadIdx.x;
  if (idx >= B_ * NB_ * NTOK) return;
  int n = idx % NTOK;
  int o = (idx / NTOK) % NB_;
  int b = idx / (NTOK * NB_);
  float acc = bias[o];
  #pragma unroll
  for (int i = 0; i < NB_; ++i)
    acc += w[o * NB_ + i] * tmpq[((size_t)b * NB_ + i) * NTOK + n];
  g[((size_t)b * NB_ + o) * NTOK + n] = 1.f / (1.f + __expf(-acc));
}

// x_gated[b,n,c] = g[b,br,n] * m_out[br,b,n,d]
__global__ void __launch_bounds__(256)
k_xgated(const float* __restrict__ g, const float* __restrict__ mout,
         float* __restrict__ xg) {
  long long idx = (long long)blockIdx.x * 256 + threadIdx.x;
  if (idx >= (long long)BN_ * C_) return;
  int c = (int)(idx & (C_ - 1));
  long long bn = idx >> 8;
  int b = (int)(bn / NTOK);
  int n = (int)(bn - (long long)b * NTOK);
  int br = c >> 6, d = c & (DM_ - 1);
  xg[idx] = g[((size_t)b * NB_ + br) * NTOK + n] *
            mout[(((size_t)br * B_ + b) * NTOK + n) * DM_ + d];
}

// depthwise conv K=3 pad=1 along N, channels-last
__global__ void __launch_bounds__(256)
k_dwconv_n(const float* __restrict__ in, int Cch, int in_ld,
           const float* __restrict__ w, const float* __restrict__ bias,
           float* __restrict__ out, int out_ld, int out_off) {
  long long idx = (long long)blockIdx.x * 256 + threadIdx.x;
  if (idx >= (long long)BN_ * Cch) return;
  int c = (int)(idx % Cch);
  long long t = idx / Cch;
  int n = (int)(t % NTOK);
  int b = (int)(t / NTOK);
  float acc = bias[c];
  #pragma unroll
  for (int k = 0; k < 3; ++k) {
    int nn = n + k - 1;
    if (nn >= 0 && nn < NTOK)
      acc += w[c * 3 + k] * in[((size_t)b * NTOK + nn) * in_ld + c];
  }
  out[((size_t)b * NTOK + n) * out_ld + out_off + c] = acc;
}

// cat = [xt | r]  (B,N,2C)
__global__ void __launch_bounds__(256)
k_cat(const float* __restrict__ xt, const float* __restrict__ r,
      float* __restrict__ cat) {
  long long idx = (long long)blockIdx.x * 256 + threadIdx.x;
  if (idx >= (long long)BN_ * 2 * C_) return;
  int c = (int)(idx & (2 * C_ - 1));
  long long m = idx >> 9;
  cat[idx] = (c < C_) ? xt[m * C_ + c] : r[m * C_ + (c - C_)];
}

// ---------------------------------------------------------------------------
// Flash attention: one wave handles a 16-query tile of one (b,h), streaming
// key/value tiles of 32 with online softmax.  S = Q*K^T and O += P*V are each
// exact v_wmma_f32_16x16x32_f16 shapes (hd = 32).
// ---------------------------------------------------------------------------
__global__ void __launch_bounds__(32)
k_attn(const float* __restrict__ qkv, float* __restrict__ xglob) {
  const int lane = threadIdx.x;
  const int qt = blockIdx.x;             // 0..143
  const int bh = blockIdx.y;             // 0..31
  const int b = bh >> 3, h = bh & 7;
  const float scale = 0.17677669529663687f;   // hd^-0.5
  const float* qp = qkv + (size_t)b * NTOK * 768 + h * HD_;
  const float* kp = qp + 256;
  const float* vp = qp + 512;
  const int nl = lane & 15;
  const int kb = (lane >> 4) << 3;
  v16h qf;
  {
    const float* qr = qp + (size_t)(qt * 16 + nl) * 768 + kb;
    float4 a0 = *(const float4*)(qr);
    float4 a1 = *(const float4*)(qr + 4);
    float4 a2 = *(const float4*)(qr + 16);
    float4 a3 = *(const float4*)(qr + 20);
    qf[0]=(_Float16)(a0.x*scale); qf[1]=(_Float16)(a0.y*scale);
    qf[2]=(_Float16)(a0.z*scale); qf[3]=(_Float16)(a0.w*scale);
    qf[4]=(_Float16)(a1.x*scale); qf[5]=(_Float16)(a1.y*scale);
    qf[6]=(_Float16)(a1.z*scale); qf[7]=(_Float16)(a1.w*scale);
    qf[8]=(_Float16)(a2.x*scale); qf[9]=(_Float16)(a2.y*scale);
    qf[10]=(_Float16)(a2.z*scale); qf[11]=(_Float16)(a2.w*scale);
    qf[12]=(_Float16)(a3.x*scale); qf[13]=(_Float16)(a3.y*scale);
    qf[14]=(_Float16)(a3.z*scale); qf[15]=(_Float16)(a3.w*scale);
  }
  v8f o0 = {}, o1 = {};
  float mrow[8], lrow[8];
  #pragma unroll
  for (int v = 0; v < 8; ++v) { mrow[v] = -1e30f; lrow[v] = 0.f; }
  __shared__ _Float16 Pld[16 * 32];
  for (int kt = 0; kt < NTOK / 32; ++kt) {
    const int key0 = kt * 32;
    v16h kf0, kf1;
    {
      const float* kr0 = kp + (size_t)(key0 + nl) * 768 + kb;
      const float* kr1 = kp + (size_t)(key0 + 16 + nl) * 768 + kb;
      float4 u0 = *(const float4*)(kr0),  u1 = *(const float4*)(kr0 + 4);
      float4 u2 = *(const float4*)(kr0+16), u3 = *(const float4*)(kr0 + 20);
      float4 w0 = *(const float4*)(kr1),  w1 = *(const float4*)(kr1 + 4);
      float4 w2 = *(const float4*)(kr1+16), w3 = *(const float4*)(kr1 + 20);
      kf0[0]=(_Float16)u0.x; kf0[1]=(_Float16)u0.y; kf0[2]=(_Float16)u0.z; kf0[3]=(_Float16)u0.w;
      kf0[4]=(_Float16)u1.x; kf0[5]=(_Float16)u1.y; kf0[6]=(_Float16)u1.z; kf0[7]=(_Float16)u1.w;
      kf0[8]=(_Float16)u2.x; kf0[9]=(_Float16)u2.y; kf0[10]=(_Float16)u2.z; kf0[11]=(_Float16)u2.w;
      kf0[12]=(_Float16)u3.x; kf0[13]=(_Float16)u3.y; kf0[14]=(_Float16)u3.z; kf0[15]=(_Float16)u3.w;
      kf1[0]=(_Float16)w0.x; kf1[1]=(_Float16)w0.y; kf1[2]=(_Float16)w0.z; kf1[3]=(_Float16)w0.w;
      kf1[4]=(_Float16)w1.x; kf1[5]=(_Float16)w1.y; kf1[6]=(_Float16)w1.z; kf1[7]=(_Float16)w1.w;
      kf1[8]=(_Float16)w2.x; kf1[9]=(_Float16)w2.y; kf1[10]=(_Float16)w2.z; kf1[11]=(_Float16)w2.w;
      kf1[12]=(_Float16)w3.x; kf1[13]=(_Float16)w3.y; kf1[14]=(_Float16)w3.z; kf1[15]=(_Float16)w3.w;
    }
    v8f s0 = {}, s1 = {};
    s0 = __builtin_amdgcn_wmma_f32_16x16x32_f16(false, qf, false, kf0, (short)0, s0, false, false);
    s1 = __builtin_amdgcn_wmma_f32_16x16x32_f16(false, qf, false, kf1, (short)0, s1, false, false);
    // online softmax: row reductions across the 16 lanes of each half
    #pragma unroll
    for (int v = 0; v < 8; ++v) {
      float rm = fmaxf(s0[v], s1[v]);
      #pragma unroll
      for (int off = 1; off < 16; off <<= 1)
        rm = fmaxf(rm, __shfl_xor(rm, off, 32));
      float mnew = fmaxf(mrow[v], rm);
      float corr = __expf(mrow[v] - mnew);
      mrow[v] = mnew;
      float p0 = __expf(s0[v] - mnew);
      float p1 = __expf(s1[v] - mnew);
      s0[v] = p0; s1[v] = p1;
      float rs = p0 + p1;
      #pragma unroll
      for (int off = 1; off < 16; off <<= 1)
        rs += __shfl_xor(rs, off, 32);
      lrow[v] = lrow[v] * corr + rs;
      o0[v] *= corr; o1[v] *= corr;
    }
    // stage P (16q x 32keys) through LDS: C-frag layout -> A-frag layout
    #pragma unroll
    for (int v = 0; v < 8; ++v) {
      int r = kb + v;                    // rows 0-7 (lanes<16) / 8-15
      Pld[r * 32 + nl]      = (_Float16)s0[v];
      Pld[r * 32 + 16 + nl] = (_Float16)s1[v];
    }
    __syncthreads();
    v16h pf;
    {
      const _Float16* pr = &Pld[nl * 32];
      #pragma unroll
      for (int j = 0; j < 8; ++j) {
        pf[j]     = pr[kb + j];
        pf[j + 8] = pr[16 + kb + j];
      }
    }
    __syncthreads();
    v16h vf0, vf1;
    #pragma unroll
    for (int j = 0; j < 8; ++j) {
      const float* vr0 = vp + (size_t)(key0 + kb + j) * 768;
      const float* vr1 = vp + (size_t)(key0 + 16 + kb + j) * 768;
      vf0[j]     = (_Float16)vr0[nl];
      vf0[j + 8] = (_Float16)vr1[nl];
      vf1[j]     = (_Float16)vr0[16 + nl];
      vf1[j + 8] = (_Float16)vr1[16 + nl];
    }
    o0 = __builtin_amdgcn_wmma_f32_16x16x32_f16(false, pf, false, vf0, (short)0, o0, false, false);
    o1 = __builtin_amdgcn_wmma_f32_16x16x32_f16(false, pf, false, vf1, (short)0, o1, false, false);
  }
  #pragma unroll
  for (int v = 0; v < 8; ++v) {
    int m = qt * 16 + kb + v;
    float inv = 1.f / lrow[v];
    float* orow = xglob + ((size_t)b * NTOK + m) * C_ + h * HD_;
    orow[nl]      = o0[v] * inv;
    orow[16 + nl] = o1[v] * inv;
  }
}

// spatial mean / inv-std per (b,c)
__global__ void __launch_bounds__(256)
k_spstats(const float* __restrict__ x, float* __restrict__ st) {
  int bc = blockIdx.x, t = threadIdx.x;
  int b = bc >> 8, c = bc & 255;
  float s = 0.f, s2 = 0.f;
  for (int n = t; n < NTOK; n += 256) {
    float v = x[((size_t)b * NTOK + n) * C_ + c];
    s += v; s2 += v * v;
  }
  __shared__ float r1[256], r2[256];
  r1[t] = s; r2[t] = s2;
  __syncthreads();
  for (int o = 128; o > 0; o >>= 1) {
    if (t < o) { r1[t] += r1[t + o]; r2[t] += r2[t + o]; }
    __syncthreads();
  }
  if (t == 0) {
    float mu = r1[0] / (float)NTOK;
    float var = r2[0] / (float)NTOK - mu * mu;
    st[bc * 2] = mu;
    st[bc * 2 + 1] = rsqrtf(var + 1e-5f);
  }
}

// spatial-norm + depthwise 3x3 conv (zero pad), channels-last in/out
__global__ void __launch_bounds__(256)
k_spdw(const float* __restrict__ x, const float* __restrict__ st,
       const float* __restrict__ w, const float* __restrict__ bias,
       float* __restrict__ out) {
  long long idx = (long long)blockIdx.x * 256 + threadIdx.x;
  if (idx >= (long long)BN_ * C_) return;
  int c = (int)(idx & (C_ - 1));
  long long bn = idx >> 8;
  int b = (int)(bn / NTOK);
  int n = (int)(bn - (long long)b * NTOK);
  int hh = n / W_, ww = n - hh * W_;
  float mu = st[(b * C_ + c) * 2];
  float rstd = st[(b * C_ + c) * 2 + 1];
  float acc = bias[c];
  #pragma unroll
  for (int kh = 0; kh < 3; ++kh) {
    int h2 = hh + kh - 1;
    if (h2 < 0 || h2 >= H_) continue;
    #pragma unroll
    for (int kw = 0; kw < 3; ++kw) {
      int w2 = ww + kw - 1;
      if (w2 < 0 || w2 >= W_) continue;
      float v = (x[((size_t)b * NTOK + h2 * W_ + w2) * C_ + c] - mu) * rstd;
      acc += w[c * 9 + kh * 3 + kw] * v;
    }
  }
  out[idx] = acc;
}

// (B,N,C) -> (B,C,N) final output
__global__ void __launch_bounds__(256)
k_transpose_out(const float* __restrict__ in, float* __restrict__ out) {
  long long idx = (long long)blockIdx.x * 256 + threadIdx.x;
  if (idx >= (long long)BN_ * C_) return;
  int c = (int)(idx & (C_ - 1));
  long long bn = idx >> 8;
  int b = (int)(bn / NTOK);
  int n = (int)(bn - (long long)b * NTOK);
  out[((size_t)b * C_ + c) * NTOK + n] = in[idx];
}

// ---------------------------------------------------------------------------
extern "C" void kernel_launch(void* const* d_in, const int* in_sizes, int n_in,
                              void* d_out, int out_size, void* d_ws, size_t ws_size,
                              hipStream_t stream) {
  (void)in_sizes; (void)n_in; (void)out_size; (void)ws_size;
  const float* x         = (const float*)d_in[0];
  const float* norm_w    = (const float*)d_in[1];
  const float* norm_b    = (const float*)d_in[2];
  const float* m_in_w    = (const float*)d_in[3];
  const float* m_conv_w  = (const float*)d_in[4];
  const float* m_conv_b  = (const float*)d_in[5];
  const float* m_xproj_w = (const float*)d_in[6];
  const float* m_dt_w    = (const float*)d_in[7];
  const float* m_dt_b    = (const float*)d_in[8];
  const float* m_Alog    = (const float*)d_in[9];
  const float* m_D       = (const float*)d_in[10];
  const float* m_out_w   = (const float*)d_in[11];
  const float* skip      = (const float*)d_in[12];
  const float* qdw_w     = (const float*)d_in[13];
  const float* qdw_b     = (const float*)d_in[14];
  const float* qpw_w     = (const float*)d_in[15];
  const float* qpw_b     = (const float*)d_in[16];
  const float* rdw_w     = (const float*)d_in[17];
  const float* rdw_b     = (const float*)d_in[18];
  const float* rpw_w     = (const float*)d_in[19];
  const float* rpw_b     = (const float*)d_in[20];
  const float* fdw_w     = (const float*)d_in[21];
  const float* fdw_b     = (const float*)d_in[22];
  const float* fpw_w     = (const float*)d_in[23];
  const float* fpw_b     = (const float*)d_in[24];
  const float* proj_w    = (const float*)d_in[25];
  const float* proj_b    = (const float*)d_in[26];
  const float* lconv_w   = (const float*)d_in[27];
  const float* lconv_b   = (const float*)d_in[28];
  const float* attn_in_w = (const float*)d_in[29];
  const float* attn_in_b = (const float*)d_in[30];
  const float* attn_out_w= (const float*)d_in[31];
  const float* attn_out_b= (const float*)d_in[32];
  const float* lgm_norm_w= (const float*)d_in[33];
  const float* lgm_norm_b= (const float*)d_in[34];
  const float* lgm_fc_w  = (const float*)d_in[35];
  const float* lgm_fc_b  = (const float*)d_in[36];
  const float* lgm_dw_w  = (const float*)d_in[37];
  const float* lgm_dw_b  = (const float*)d_in[38];

  float* Wf = (float*)d_ws;
  // arena (floats); aliasing is lifetime-checked
  float* XT     = Wf + 0;                 // 2359296
  float* XNORM  = Wf + 2359296;           // 2359296
  float* XZ     = Wf + 4718592;           // 9437184  (later: QKV)
  float* XM     = Wf + 14155776;          // 4718592  (later: XGLOB)
  float* XPROJ  = Wf + 18874368;          // 1327104  (later: TMPQ,G)
  float* DT     = Wf + 20201472;          // 4718592  (later: CAT2)
  float* Y      = Wf + 24920064;          // 4718592  (later: XOUT)
  float* MOUT   = Wf + 29638656;          // 2359296  (later: YDW)
  float* XCAT   = Wf + 31997952;          // 2359296
  float* XGATED = Wf + 34357248;          // 2359296
  float* TMP1   = Wf + 36716544;          // 2359296  (later: FBUF)
  float* RBUF   = Wf + 39075840;          // 2359296  (later: PBUF)
  float* CAT    = Wf + 41435136;          // 4718592
  float* TMP2   = Wf + 46153728;          // 4718592
  float* XMSEQ  = Wf + 50872320;          // 2359296
  float* SSTAT  = Wf + 53231616;          // 2048
  float* QKV = XZ;  float* XGLOB = XM;  float* TMPQ = XPROJ;
  float* Gg = XPROJ + 36864;  float* CAT2 = DT;  float* XOUT = Y;
  float* YDW = MOUT;  float* FBUF = TMP1;  float* PBUF = RBUF;
  float* OUT = (float*)d_out;

  #define EW(total) dim3((unsigned)(((long long)(total) + 255) / 256)), dim3(256)

  // 1. shared LayerNorm
  k_ln_nchw<<<dim3(BN_), dim3(256), 0, stream>>>(x, norm_w, norm_b, XT, XNORM);
  // 2. mamba in_proj (batched over branches): (9216,64)x(256,64)^T
  k_gemm<<<dim3(4, BN_/32, NB_), dim3(32), 0, stream>>>(
      XNORM, C_, (long long)DM_, m_in_w, (long long)(2*DI_*DM_),
      nullptr, nullptr, 0, XZ, 2*DI_, 0, (long long)BN_*2*DI_, BN_, 2*DI_, DM_);
  // 3. causal dwconv + SiLU
  k_mamba_conv<<<EW((long long)NB_*BN_*DI_), 0, stream>>>(XZ, m_conv_w, m_conv_b, XM);
  // 4. x_proj: (9216,128)x(36,128)^T
  k_gemm<<<dim3(1, BN_/32, NB_), dim3(32), 0, stream>>>(
      XM, DI_, (long long)BN_*DI_, m_xproj_w, (long long)(36*DI_),
      nullptr, nullptr, 0, XPROJ, 36, 0, (long long)BN_*36, BN_, 36, DI_);
  // 5. dt projection + softplus
  k_dtproj<<<EW((long long)NB_*BN_*DI_), 0, stream>>>(XPROJ, m_dt_w, m_dt_b, DT);
  // 6. selective scan
  k_scan<<<dim3(8), dim3(256), 0, stream>>>(DT, XM, XPROJ, m_Alog, m_D, Y);
  // 7a. y *= silu(z)
  k_ygate<<<EW((long long)NB_*BN_*DI_), 0, stream>>>(Y, XZ);
  // 7b. out_proj: (9216,128)x(64,128)^T
  k_gemm<<<dim3(1, BN_/32, NB_), dim3(32), 0, stream>>>(
      Y, DI_, (long long)BN_*DI_, m_out_w, (long long)(DM_*DI_),
      nullptr, nullptr, 0, MOUT, DM_, 0, (long long)BN_*DM_, BN_, DM_, DI_);
  // 7c. skip + scatter into x_concat
  k_mout_skip<<<EW((long long)NB_*BN_*DM_), 0, stream>>>(MOUT, XNORM, skip, XCAT);
  // 8. gating
  k_qdw<<<EW(B_*NB_*NTOK), 0, stream>>>(XCAT, qdw_w, qdw_b, TMPQ);
  k_qpw<<<EW(B_*NB_*NTOK), 0, stream>>>(TMPQ, qpw_w, qpw_b, Gg);
  // 9. x_gated
  k_xgated<<<EW((long long)BN_*C_), 0, stream>>>(Gg, MOUT, XGATED);
  // 10. R = pw(dw(xg)) + xg
  k_dwconv_n<<<EW((long long)BN_*C_), 0, stream>>>(XGATED, C_, C_, rdw_w, rdw_b, TMP1, C_, 0);
  k_gemm<<<dim3(4, BN_/32, 1), dim3(32), 0, stream>>>(
      TMP1, C_, 0, rpw_w, 0, rpw_b, XGATED, C_, RBUF, C_, 0, 0, BN_, C_, C_);
  // 11. cat = [xt | R]
  k_cat<<<EW((long long)BN_*2*C_), 0, stream>>>(XT, RBUF, CAT);
  // 12-13. xm = pw(dw(cat)) + cat[:, :C]
  k_dwconv_n<<<EW((long long)BN_*2*C_), 0, stream>>>(CAT, 2*C_, 2*C_, fdw_w, fdw_b, TMP2, 2*C_, 0);
  k_gemm<<<dim3(4, BN_/32, 1), dim3(32), 0, stream>>>(
      TMP2, 2*C_, 0, fpw_w, 0, fpw_b, CAT, 2*C_, XMSEQ, C_, 0, 0, BN_, C_, 2*C_);
  // 14. x_local -> cat2[:, :C]
  k_dwconv_n<<<EW((long long)BN_*C_), 0, stream>>>(XMSEQ, C_, C_, lconv_w, lconv_b, CAT2, 2*C_, 0);
  // 15. qkv
  k_gemm<<<dim3(12, BN_/32, 1), dim3(32), 0, stream>>>(
      XMSEQ, C_, 0, attn_in_w, 0, attn_in_b, nullptr, 0, QKV, 3*C_, 0, 0, BN_, 3*C_, C_);
  // 16. flash attention
  k_attn<<<dim3(NTOK/16, B_*NH_), dim3(32), 0, stream>>>(QKV, XGLOB);
  // 17. attn out proj -> cat2[:, C:]
  k_gemm<<<dim3(4, BN_/32, 1), dim3(32), 0, stream>>>(
      XGLOB, C_, 0, attn_out_w, 0, attn_out_b, nullptr, 0, CAT2, 2*C_, C_, 0, BN_, C_, C_);
  // 18. lgm fc
  k_gemm<<<dim3(4, BN_/32, 1), dim3(32), 0, stream>>>(
      CAT2, 2*C_, 0, lgm_fc_w, 0, lgm_fc_b, nullptr, 0, XOUT, C_, 0, 0, BN_, C_, 2*C_);
  // 19. LN + GeLU
  k_ln_gelu<<<dim3(BN_), dim3(256), 0, stream>>>(XOUT, lgm_norm_w, lgm_norm_b);
  // 20. spatial norm + depthwise 3x3
  k_spstats<<<dim3(B_*C_), dim3(256), 0, stream>>>(XOUT, SSTAT);
  k_spdw<<<EW((long long)BN_*C_), 0, stream>>>(XOUT, SSTAT, lgm_dw_w, lgm_dw_b, YDW);
  // 21. final shared LN
  k_ln_chlast<<<dim3(BN_), dim3(256), 0, stream>>>(YDW, norm_w, norm_b, FBUF);
  // 22. final projection
  k_gemm<<<dim3(4, BN_/32, 1), dim3(32), 0, stream>>>(
      FBUF, C_, 0, proj_w, 0, proj_b, nullptr, 0, PBUF, C_, 0, 0, BN_, C_, C_);
  // 23. (B,N,C) -> (B,C,H,W)
  k_transpose_out<<<EW((long long)BN_*C_), 0, stream>>>(PBUF, OUT);
  #undef EW
}
